// ViG_44573170598128
// MI455X (gfx1250) — matrix operand order repaired and checked by reading, not compiled, not run on hardware
//
#include <hip/hip_runtime.h>
#include <hip/hip_bf16.h>
#include <math.h>

typedef __bf16 bf16_t;
typedef __attribute__((ext_vector_type(16))) __bf16 v16bf;
typedef __attribute__((ext_vector_type(8)))  __bf16 v8bf;
typedef __attribute__((ext_vector_type(8)))  float  v8f;

#define DV     256
#define NBLK   12
#define BBATCH 2
#define NPB    2500            // tokens per batch (50x50)
#define NROWS  (BBATCH*NPB)    // 5000
#define KMAX   18
#define LDSIM  2512            // padded similarity row stride (multiple of 16)

// ------------------------------ device helpers ------------------------------

__device__ __forceinline__ float gelu_f(float v) {
  return 0.5f * v * (1.0f + erff(v * 0.70710678118654752440f));
}

// A-operand fragment (16x32 MxK, 16-bit), p already points at row base + lh*16:
// elements 0..15 = K[+0..16)
__device__ __forceinline__ v16bf ldA16(const bf16_t* __restrict__ p) {
  v8bf a = *reinterpret_cast<const v8bf*>(p);
  v8bf b = *reinterpret_cast<const v8bf*>(p + 8);
  v16bf f;
#pragma unroll
  for (int i = 0; i < 8; ++i) { f[i] = a[i]; f[8 + i] = b[i]; }
  return f;
}

// B-operand fragment (32x16 KxN, 16-bit), p already points at row base + lh*16:
// elements 0..15 = K[+0..16)
__device__ __forceinline__ v16bf ldB16(const bf16_t* __restrict__ p) {
  v8bf a = *reinterpret_cast<const v8bf*>(p);
  v8bf b = *reinterpret_cast<const v8bf*>(p + 8);
  v16bf f;
#pragma unroll
  for (int i = 0; i < 8; ++i) { f[i] = a[i]; f[8 + i] = b[i]; }
  return f;
}

#define WMMA_BF16(A_, B_, C_) \
  __builtin_amdgcn_wmma_f32_16x16x32_bf16(false, (A_), false, (B_), (short)0, (C_), false, false)

// ------------------------------ GEMM kernels ------------------------------
// out[n, o] = sum_k A[o, k] * Bm[n, k]   (A: weights MxK, Bm: activations RxK)
// Per-wave tile: 32(M) x 64(N); block = 4 waves => 32 x 256 tile.
// OOB rows are CLAMPED (not zeroed): an OOB A row only corrupts D rows that
// are never stored, an OOB B row only corrupts D columns never stored, so the
// k-loop carries no predication at all.
__global__ void __launch_bounds__(128) gemm_nc_k(
    const bf16_t* __restrict__ A, const bf16_t* __restrict__ Bm,
    bf16_t* __restrict__ out,
    const float* __restrict__ bias, const float* __restrict__ gam,
    const float* __restrict__ bet, const bf16_t* __restrict__ resid,
    int M, int K, long R, int do_gelu)
{
  const int lane = (int)(threadIdx.x & 31u);
  const int wv   = (int)(threadIdx.x >> 5);
  const int lh   = lane >> 4;
  const int lr   = lane & 15;
  const long n0  = (long)blockIdx.x * 256 + wv * 64;
  const int  m0  = (int)blockIdx.y * 32;

  long ar0 = m0 + lr;      if (ar0 >= M) ar0 = M - 1;
  long ar1 = m0 + 16 + lr; if (ar1 >= M) ar1 = M - 1;
  const bf16_t* aP0 = A + ar0 * (long)K + lh * 16;
  const bf16_t* aP1 = A + ar1 * (long)K + lh * 16;
  long br0 = n0 + lr;      if (br0 >= R) br0 = R - 1;
  long br1 = n0 + 16 + lr; if (br1 >= R) br1 = R - 1;
  long br2 = n0 + 32 + lr; if (br2 >= R) br2 = R - 1;
  long br3 = n0 + 48 + lr; if (br3 >= R) br3 = R - 1;
  const bf16_t* bP0 = Bm + br0 * (long)K + lh * 16;
  const bf16_t* bP1 = Bm + br1 * (long)K + lh * 16;
  const bf16_t* bP2 = Bm + br2 * (long)K + lh * 16;
  const bf16_t* bP3 = Bm + br3 * (long)K + lh * 16;

  v8f acc[2][4];
#pragma unroll
  for (int mi = 0; mi < 2; ++mi)
#pragma unroll
    for (int ni = 0; ni < 4; ++ni)
#pragma unroll
      for (int e = 0; e < 8; ++e) acc[mi][ni][e] = 0.f;

  for (int k0 = 0; k0 < K; k0 += 32) {
    v16bf a0 = ldA16(aP0 + k0);
    v16bf a1 = ldA16(aP1 + k0);
    v16bf b0 = ldB16(bP0 + k0);
    v16bf b1 = ldB16(bP1 + k0);
    v16bf b2 = ldB16(bP2 + k0);
    v16bf b3 = ldB16(bP3 + k0);
    acc[0][0] = WMMA_BF16(a0, b0, acc[0][0]);
    acc[0][1] = WMMA_BF16(a0, b1, acc[0][1]);
    acc[0][2] = WMMA_BF16(a0, b2, acc[0][2]);
    acc[0][3] = WMMA_BF16(a0, b3, acc[0][3]);
    acc[1][0] = WMMA_BF16(a1, b0, acc[1][0]);
    acc[1][1] = WMMA_BF16(a1, b1, acc[1][1]);
    acc[1][2] = WMMA_BF16(a1, b2, acc[1][2]);
    acc[1][3] = WMMA_BF16(a1, b3, acc[1][3]);
  }

#pragma unroll
  for (int ni = 0; ni < 4; ++ni) {
    long n = n0 + ni * 16 + lr;
    if (n >= R) continue;
#pragma unroll
    for (int mi = 0; mi < 2; ++mi) {
      v8f accv = acc[mi][ni];
      int ob = m0 + mi * 16 + lh * 8;
      v8bf ov;
#pragma unroll
      for (int e = 0; e < 8; ++e) {
        int o = ob + e;
        float v = accv[e];
        if (bias) v += bias[o];
        if (gam)  v = v * gam[o] + bet[o];
        if (do_gelu) v = gelu_f(v);
        if (resid) v += (float)resid[n * (long)M + o];
        ov[e] = (bf16_t)v;
      }
      *reinterpret_cast<v8bf*>(out + n * (long)M + ob) = ov;
    }
  }
}

// Similarity: S[n, m] = sum_c Xn[m, c] * Xn[n, c]  per batch (blockIdx.z)
__global__ void __launch_bounds__(128) gemm_sim_k(
    const bf16_t* __restrict__ Xn, float* __restrict__ S)
{
  const int lane = (int)(threadIdx.x & 31u);
  const int wv   = (int)(threadIdx.x >> 5);
  const int lh   = lane >> 4;
  const int lr   = lane & 15;
  const long n0  = (long)blockIdx.x * 256 + wv * 64;
  const int  m0  = (int)blockIdx.y * 32;
  const bf16_t* Ab = Xn + (long)blockIdx.z * NPB * DV;
  float* Sb = S + (long)blockIdx.z * NPB * LDSIM;

  long ar0 = m0 + lr;      if (ar0 >= NPB) ar0 = NPB - 1;
  long ar1 = m0 + 16 + lr; if (ar1 >= NPB) ar1 = NPB - 1;
  const bf16_t* aP0 = Ab + ar0 * (long)DV + lh * 16;
  const bf16_t* aP1 = Ab + ar1 * (long)DV + lh * 16;
  long br0 = n0 + lr;      if (br0 >= NPB) br0 = NPB - 1;
  long br1 = n0 + 16 + lr; if (br1 >= NPB) br1 = NPB - 1;
  long br2 = n0 + 32 + lr; if (br2 >= NPB) br2 = NPB - 1;
  long br3 = n0 + 48 + lr; if (br3 >= NPB) br3 = NPB - 1;
  const bf16_t* bP0 = Ab + br0 * (long)DV + lh * 16;
  const bf16_t* bP1 = Ab + br1 * (long)DV + lh * 16;
  const bf16_t* bP2 = Ab + br2 * (long)DV + lh * 16;
  const bf16_t* bP3 = Ab + br3 * (long)DV + lh * 16;

  v8f acc[2][4];
#pragma unroll
  for (int mi = 0; mi < 2; ++mi)
#pragma unroll
    for (int ni = 0; ni < 4; ++ni)
#pragma unroll
      for (int e = 0; e < 8; ++e) acc[mi][ni][e] = 0.f;

  for (int k0 = 0; k0 < DV; k0 += 32) {
    v16bf a0 = ldA16(aP0 + k0);
    v16bf a1 = ldA16(aP1 + k0);
    v16bf b0 = ldB16(bP0 + k0);
    v16bf b1 = ldB16(bP1 + k0);
    v16bf b2 = ldB16(bP2 + k0);
    v16bf b3 = ldB16(bP3 + k0);
    acc[0][0] = WMMA_BF16(a0, b0, acc[0][0]);
    acc[0][1] = WMMA_BF16(a0, b1, acc[0][1]);
    acc[0][2] = WMMA_BF16(a0, b2, acc[0][2]);
    acc[0][3] = WMMA_BF16(a0, b3, acc[0][3]);
    acc[1][0] = WMMA_BF16(a1, b0, acc[1][0]);
    acc[1][1] = WMMA_BF16(a1, b1, acc[1][1]);
    acc[1][2] = WMMA_BF16(a1, b2, acc[1][2]);
    acc[1][3] = WMMA_BF16(a1, b3, acc[1][3]);
  }

#pragma unroll
  for (int ni = 0; ni < 4; ++ni) {
    long n = n0 + ni * 16 + lr;
    if (n >= NPB) continue;
#pragma unroll
    for (int mi = 0; mi < 2; ++mi) {
      v8f accv = acc[mi][ni];
      int ob = m0 + mi * 16 + lh * 8;
#pragma unroll
      for (int e = 0; e < 8; ++e) {
        int m = ob + e;
        if (m < NPB) Sb[n * (long)LDSIM + m] = accv[e];
      }
    }
  }
}

// ------------------------------ stem kernels ------------------------------

__global__ void __launch_bounds__(256) conv3x3_k(
    const float* __restrict__ in, const float* __restrict__ w,
    const float* __restrict__ bias, float* __restrict__ out,
    int Bn, int Ci, int Hi, int Wi, int Co, int Ho, int Wo, int stride)
{
  long tid = (long)blockIdx.x * 256 + threadIdx.x;
  long total = (long)Bn * Co * Ho * Wo;
  if (tid >= total) return;
  int x = (int)(tid % Wo); long t = tid / Wo;
  int y = (int)(t % Ho);   t /= Ho;
  int o = (int)(t % Co);
  int b = (int)(t / Co);
  float acc = bias[o];
  const float* wp = w + (long)o * Ci * 9;
  for (int ci = 0; ci < Ci; ++ci) {
    const float* ip = in + ((long)(b * Ci + ci) * Hi) * Wi;
    const float* wc = wp + ci * 9;
#pragma unroll
    for (int ky = 0; ky < 3; ++ky) {
      int iy = y * stride + ky - 1;
      if (iy < 0 || iy >= Hi) continue;
#pragma unroll
      for (int kx = 0; kx < 3; ++kx) {
        int ix = x * stride + kx - 1;
        if (ix < 0 || ix >= Wi) continue;
        acc += ip[(long)iy * Wi + ix] * wc[ky * 3 + kx];
      }
    }
  }
  out[tid] = acc;
}

// in-place GroupNorm(32) (+ optional exact GELU); one block per (group, batch)
__global__ void __launch_bounds__(256) groupnorm_k(
    float* __restrict__ x, const float* __restrict__ g,
    const float* __restrict__ be, int C, long HW, int do_gelu)
{
  int grp = (int)blockIdx.x;
  int b   = (int)blockIdx.y;
  int cpg = C / 32;
  float* base = x + ((long)b * C + (long)grp * cpg) * HW;
  long n = (long)cpg * HW;
  float s = 0.f, ss = 0.f;
  for (long i = threadIdx.x; i < n; i += 256) {
    float v = base[i]; s += v; ss += v * v;
  }
  __shared__ float rs[256], rq[256];
  rs[threadIdx.x] = s; rq[threadIdx.x] = ss;
  __syncthreads();
  for (int st = 128; st > 0; st >>= 1) {
    if ((int)threadIdx.x < st) {
      rs[threadIdx.x] += rs[threadIdx.x + st];
      rq[threadIdx.x] += rq[threadIdx.x + st];
    }
    __syncthreads();
  }
  float mean = rs[0] / (float)n;
  float var  = rq[0] / (float)n - mean * mean;
  float rstd = rsqrtf(var + 1e-5f);
  for (long i = threadIdx.x; i < n; i += 256) {
    long c = (long)grp * cpg + i / HW;
    float v = (base[i] - mean) * rstd * g[c] + be[c];
    if (do_gelu) v = gelu_f(v);
    base[i] = v;
  }
}

// s4 [B,256,2500] fp32 + DETR sine pos-embed -> bf16 token-major [B*2500, 256]
__global__ void __launch_bounds__(256) posembed_pack_k(
    const float* __restrict__ s4, bf16_t* __restrict__ xo)
{
  long n = blockIdx.x;
  int batch = (int)(n / NPB);
  int nl = (int)(n % NPB);
  int c = (int)threadIdx.x;
  int y = nl / 50, xx = nl % 50;
  const float twopi = 6.2831853071795864769f;
  const int hlf = DV / 2;
  int j = (c < hlf) ? c : c - hlf;
  float coord = (c < hlf) ? ((float)(y + 1) / 50.f * twopi)
                          : ((float)(xx + 1) / 50.f * twopi);
  int t = j >> 1;
  float dt = powf(10000.f, 2.f * (float)t / (float)hlf);
  float a = coord / dt;
  float pos = (j & 1) ? cosf(a) : sinf(a);
  float v = s4[((long)(batch * DV + c)) * NPB + nl] + pos;
  xo[n * DV + c] = (bf16_t)v;
}

// ------------------------------ graph kernels ------------------------------

// per-row channel L2 normalize (eps 1e-12), bf16 -> bf16
__global__ void __launch_bounds__(256) l2norm_k(
    const bf16_t* __restrict__ h, bf16_t* __restrict__ xn)
{
  long row = blockIdx.x;
  int c = (int)threadIdx.x;
  float v = (float)h[row * DV + c];
  __shared__ float red[256];
  red[c] = v * v;
  __syncthreads();
  for (int st = 128; st > 0; st >>= 1) {
    if (c < st) red[c] += red[c + st];
    __syncthreads();
  }
  float r = rsqrtf(red[0] + 1e-12f);
  xn[row * DV + c] = (bf16_t)(v * r);
}

// iterative wave-argmax top-(k*dil) per row (LDS-resident), keep every dil-th
__global__ void __launch_bounds__(32) topk_k(
    const float* __restrict__ S, int* __restrict__ idx, int k, int dil)
{
  __shared__ float sm[NPB];
  int row = (int)blockIdx.x;
  int batch = (int)blockIdx.y;
  const float* Sr = S + ((long)batch * NPB + row) * LDSIM;
  int lane = (int)threadIdx.x;
  for (int i = lane; i < NPB; i += 32) sm[i] = Sr[i];
  __syncthreads();
  int kk = k * dil;
  for (int j = 0; j < kk; ++j) {
    float bv = -3.0e38f; int bi = 0;
    for (int i = lane; i < NPB; i += 32) {
      float v = sm[i];
      if (v > bv) { bv = v; bi = i; }
    }
    for (int off = 16; off > 0; off >>= 1) {
      float ov = __shfl_xor(bv, off);
      int   oi = __shfl_xor(bi, off);
      if (ov > bv || (ov == bv && oi < bi)) { bv = ov; bi = oi; }
    }
    if (lane == 0) {
      if (j % dil == 0) idx[((long)batch * NPB + row) * KMAX + j / dil] = bi;
      sm[bi] = -3.0e38f;
    }
    __syncthreads();
  }
}

// F[n, 2c] = H[n,c]; F[n, 2c+1] = max_j H[idx[n,j], c] - H[n,c]
__global__ void __launch_bounds__(256) gather_k(
    const bf16_t* __restrict__ H, const int* __restrict__ idx,
    bf16_t* __restrict__ F, int k)
{
  long n = blockIdx.x;
  int c = (int)threadIdx.x;
  int batch = (int)(n / NPB);
  long bbase = (long)batch * NPB * DV;
  float h = (float)H[n * DV + c];
  const int* id = idx + n * KMAX;
  float m = -3.0e38f;
  for (int j = 0; j < k; ++j) {
    int nb = id[j];
    float v = (float)H[bbase + (long)nb * DV + c];
    m = fmaxf(m, v);
  }
  F[n * (2 * DV) + 2 * c]     = (bf16_t)h;
  F[n * (2 * DV) + 2 * c + 1] = (bf16_t)(m - h);
}

// ------------------------------ glue kernels ------------------------------

__global__ void __launch_bounds__(256) f32_to_bf16_k(
    const float* __restrict__ s, bf16_t* __restrict__ d, long n)
{
  long i = (long)blockIdx.x * 256 + threadIdx.x;
  if (i < n) d[i] = (bf16_t)s[i];
}

// token-major bf16 [B*2500, 256] -> fp32 [B, 256, 2500] output
__global__ void __launch_bounds__(256) unpack_k(
    const bf16_t* __restrict__ x, float* __restrict__ out)
{
  long n = blockIdx.x;
  int c = (int)threadIdx.x;
  int batch = (int)(n / NPB);
  int nl = (int)(n % NPB);
  out[((long)(batch * DV + c)) * NPB + nl] = (float)x[n * DV + c];
}

// ------------------------------ host launcher ------------------------------

extern "C" void kernel_launch(void* const* d_in, const int* in_sizes, int n_in,
                              void* d_out, int out_size, void* d_ws, size_t ws_size,
                              hipStream_t stream)
{
  (void)in_sizes; (void)n_in; (void)out_size; (void)ws_size;
  static const int KS[NBLK]   = {9, 9, 10, 11, 12, 13, 13, 14, 15, 16, 17, 18};
  static const int DILS[NBLK] = {1, 1, 1, 1, 2, 2, 2, 2, 3, 3, 3, 3};

  const float* x_in = (const float*)d_in[0];
  const float *sw[5], *sb[5], *sg[5], *sbe[5];
  for (int i = 0; i < 5; ++i) {
    sw[i]  = (const float*)d_in[1 + 4 * i];
    sb[i]  = (const float*)d_in[2 + 4 * i];
    sg[i]  = (const float*)d_in[3 + 4 * i];
    sbe[i] = (const float*)d_in[4 + 4 * i];
  }
  const float* fc1_w  = (const float*)d_in[21];
  const float* fc1_b  = (const float*)d_in[22];
  const float* fc1_g  = (const float*)d_in[23];
  const float* fc1_be = (const float*)d_in[24];
  const float* mr_w   = (const float*)d_in[25];
  const float* mr_g   = (const float*)d_in[26];
  const float* mr_be  = (const float*)d_in[27];
  const float* fc2_w  = (const float*)d_in[28];
  const float* fc2_b  = (const float*)d_in[29];
  const float* fc2_g  = (const float*)d_in[30];
  const float* fc2_be = (const float*)d_in[31];
  const float* ffn1_w  = (const float*)d_in[32];
  const float* ffn1_b  = (const float*)d_in[33];
  const float* ffn1_g  = (const float*)d_in[34];
  const float* ffn1_be = (const float*)d_in[35];
  const float* ffn2_w  = (const float*)d_in[36];
  const float* ffn2_b  = (const float*)d_in[37];
  const float* ffn2_g  = (const float*)d_in[38];
  const float* ffn2_be = (const float*)d_in[39];

  // ---- workspace layout (bump allocator; stem aliases the block scratch) ----
  char* base = (char*)d_ws;
  size_t off = 0;
  auto carve = [&](size_t bytes) -> char* {
    char* p = base + off;
    off += (bytes + 255) & ~(size_t)255;
    return p;
  };
  bf16_t* wfc1  = (bf16_t*)carve((size_t)NBLK * 256 * 256 * 2);
  bf16_t* wmr   = (bf16_t*)carve((size_t)NBLK * 512 * 512 * 2);
  bf16_t* wfc2  = (bf16_t*)carve((size_t)NBLK * 256 * 512 * 2);
  bf16_t* wffn1 = (bf16_t*)carve((size_t)NBLK * 1024 * 256 * 2);
  bf16_t* wffn2 = (bf16_t*)carve((size_t)NBLK * 256 * 1024 * 2);
  bf16_t* xA    = (bf16_t*)carve((size_t)NROWS * DV * 2);
  bf16_t* xB    = (bf16_t*)carve((size_t)NROWS * DV * 2);
  char* region = base + off;   // reused between stem phase and block phase
  size_t roff = 0;
  auto rcarve = [&](size_t bytes) -> char* {
    char* p = region + roff;
    roff += (bytes + 255) & ~(size_t)255;
    return p;
  };
  bf16_t* Hbuf  = (bf16_t*)rcarve((size_t)NROWS * DV * 2);
  bf16_t* XnB   = (bf16_t*)rcarve((size_t)NROWS * DV * 2);
  bf16_t* Fbuf  = (bf16_t*)rcarve((size_t)NROWS * 2 * DV * 2);
  bf16_t* F2buf = (bf16_t*)rcarve((size_t)NROWS * 2 * DV * 2);
  bf16_t* FFNb  = (bf16_t*)rcarve((size_t)NROWS * 4 * DV * 2);
  float*  Sbuf  = (float*)rcarve((size_t)BBATCH * NPB * LDSIM * 4);
  int*    idxb  = (int*)rcarve((size_t)NROWS * KMAX * 4);
  // stem aliases (used strictly before the block phase writes the region)
  float* convA = (float*)region;
  float* convB = (float*)(region + (((size_t)2 * 32 * 400 * 400 * 4 + 255) & ~(size_t)255));

  // ---- weight conversion fp32 -> bf16 ----
  {
    long n;
    n = (long)NBLK * 256 * 256;  f32_to_bf16_k<<<dim3((unsigned)((n + 255) / 256)), 256, 0, stream>>>(fc1_w,  wfc1,  n);
    n = (long)NBLK * 512 * 512;  f32_to_bf16_k<<<dim3((unsigned)((n + 255) / 256)), 256, 0, stream>>>(mr_w,   wmr,   n);
    n = (long)NBLK * 256 * 512;  f32_to_bf16_k<<<dim3((unsigned)((n + 255) / 256)), 256, 0, stream>>>(fc2_w,  wfc2,  n);
    n = (long)NBLK * 1024 * 256; f32_to_bf16_k<<<dim3((unsigned)((n + 255) / 256)), 256, 0, stream>>>(ffn1_w, wffn1, n);
    n = (long)NBLK * 256 * 1024; f32_to_bf16_k<<<dim3((unsigned)((n + 255) / 256)), 256, 0, stream>>>(ffn2_w, wffn2, n);
  }

  // ---- stem: 5x (conv3x3 + GroupNorm(32) [+GELU on first 4]) ----
  {
    long t;
    t = (long)2 * 32 * 400 * 400;
    conv3x3_k<<<dim3((unsigned)((t + 255) / 256)), 256, 0, stream>>>(x_in, sw[0], sb[0], convA, 2, 3, 800, 800, 32, 400, 400, 2);
    groupnorm_k<<<dim3(32, 2), 256, 0, stream>>>(convA, sg[0], sbe[0], 32, 160000L, 1);
    t = (long)2 * 64 * 200 * 200;
    conv3x3_k<<<dim3((unsigned)((t + 255) / 256)), 256, 0, stream>>>(convA, sw[1], sb[1], convB, 2, 32, 400, 400, 64, 200, 200, 2);
    groupnorm_k<<<dim3(32, 2), 256, 0, stream>>>(convB, sg[1], sbe[1], 64, 40000L, 1);
    t = (long)2 * 128 * 100 * 100;
    conv3x3_k<<<dim3((unsigned)((t + 255) / 256)), 256, 0, stream>>>(convB, sw[2], sb[2], convA, 2, 64, 200, 200, 128, 100, 100, 2);
    groupnorm_k<<<dim3(32, 2), 256, 0, stream>>>(convA, sg[2], sbe[2], 128, 10000L, 1);
    t = (long)2 * 256 * 50 * 50;
    conv3x3_k<<<dim3((unsigned)((t + 255) / 256)), 256, 0, stream>>>(convA, sw[3], sb[3], convB, 2, 128, 100, 100, 256, 50, 50, 2);
    groupnorm_k<<<dim3(32, 2), 256, 0, stream>>>(convB, sg[3], sbe[3], 256, 2500L, 1);
    conv3x3_k<<<dim3((unsigned)((t + 255) / 256)), 256, 0, stream>>>(convB, sw[4], sb[4], convA, 2, 256, 50, 50, 256, 50, 50, 1);
    groupnorm_k<<<dim3(32, 2), 256, 0, stream>>>(convA, sg[4], sbe[4], 256, 2500L, 0);
  }

  // ---- pos-embed + pack to token-major bf16 ----
  posembed_pack_k<<<dim3(NROWS), 256, 0, stream>>>(convA, xA);

  // ---- 12 Grapher + FFN blocks ----
  const dim3 gemmBlk(128);
  const dim3 grid256(20, 8);     // R=5000 rows / 256-col block tile, M=256
  const dim3 grid512(20, 16);    // M=512
  const dim3 grid1024(20, 32);   // M=1024
  const dim3 gridSim(10, 79, 2); // R=M=2500 per batch
  bf16_t* xCur = xA;
  bf16_t* xNxt = xB;
  for (int i = 0; i < NBLK; ++i) {
    int k = KS[i], dil = DILS[i];
    // fc1: hfeat = affine(W1 x + b1)
    gemm_nc_k<<<grid256, gemmBlk, 0, stream>>>(
        wfc1 + (long)i * 256 * 256, xCur, Hbuf,
        fc1_b + i * 256, fc1_g + i * 256, fc1_be + i * 256, nullptr,
        256, 256, (long)NROWS, 0);
    // knn graph: normalize -> NxN similarity -> top-k
    l2norm_k<<<dim3(NROWS), 256, 0, stream>>>(Hbuf, XnB);
    gemm_sim_k<<<gridSim, gemmBlk, 0, stream>>>(XnB, Sbuf);
    topk_k<<<dim3(NPB, BBATCH), 32, 0, stream>>>(Sbuf, idxb, k, dil);
    // gather neighbors, max-diff, channel-interleave concat -> 512 ch
    gather_k<<<dim3(NROWS), 256, 0, stream>>>(Hbuf, idxb, Fbuf, k);
    // mr: gelu(affine(Wmr f))
    gemm_nc_k<<<grid512, gemmBlk, 0, stream>>>(
        wmr + (long)i * 512 * 512, Fbuf, F2buf,
        nullptr, mr_g + i * 512, mr_be + i * 512, nullptr,
        512, 512, (long)NROWS, 1);
    // fc2: x = affine(W2 f + b2) + res
    gemm_nc_k<<<grid256, gemmBlk, 0, stream>>>(
        wfc2 + (long)i * 256 * 512, F2buf, xNxt,
        fc2_b + i * 256, fc2_g + i * 256, fc2_be + i * 256, xCur,
        256, 512, (long)NROWS, 0);
    { bf16_t* t = xCur; xCur = xNxt; xNxt = t; }
    // ffn1: gelu(affine(Wf1 x + b))
    gemm_nc_k<<<grid1024, gemmBlk, 0, stream>>>(
        wffn1 + (long)i * 1024 * 256, xCur, FFNb,
        ffn1_b + i * 1024, ffn1_g + i * 1024, ffn1_be + i * 1024, nullptr,
        1024, 256, (long)NROWS, 1);
    // ffn2: x = affine(Wf2 h + b) + res
    gemm_nc_k<<<grid256, gemmBlk, 0, stream>>>(
        wffn2 + (long)i * 256 * 1024, FFNb, xNxt,
        ffn2_b + i * 256, ffn2_g + i * 256, ffn2_be + i * 256, xCur,
        256, 1024, (long)NROWS, 0);
    { bf16_t* t = xCur; xCur = xNxt; xNxt = t; }
  }

  // ---- unpack to fp32 [B, 256, 50, 50] ----
  unpack_k<<<dim3(NROWS), 256, 0, stream>>>(xCur, (float*)d_out);
}
